// GPO_17119739641993
// MI455X (gfx1250) — matrix operand order repaired and checked by reading, not compiled, not run on hardware
//
#include <hip/hip_runtime.h>
#include <math.h>

// ---------------- problem constants ----------------
#define B_    8
#define NC_   512
#define NT_   512
#define S_    1024      // NC + NT
#define DX_   1024
#define D_    512
#define DFF_  2048
#define H_    8
#define DH_   64
#define L_    4
#define NSEG_ 128
#define KIN_  1025      // DX + DY
#define KP_   1088      // KIN padded to multiple of 64
#define NEGB  (-1000000000.0f)
#define LN2PI_ 1.8378770664093453f

typedef __attribute__((ext_vector_type(16))) _Float16 v16h;
typedef __attribute__((ext_vector_type(8)))  _Float16 v8h;
typedef __attribute__((ext_vector_type(8)))  float    v8f;

// ---------------- WMMA fragment helpers ----------------
// A operand, 16x32 f16 (M x K). Lane l<16: M=l, K = kofs+0..15;
// lane l>=16: M=l-16, K = kofs+16..31.
static __device__ __forceinline__ v16h frag_a(const _Float16* base, int stride,
                                              int lane, int kofs) {
  int r  = lane & 15;
  int kh = (lane >> 4) * 16;
  const _Float16* p = base + r * stride + kofs + kh;
  v8h lo = *(const v8h*)(p);
  v8h hi = *(const v8h*)(p + 8);
  return __builtin_shufflevector(lo, hi, 0,1,2,3,4,5,6,7,8,9,10,11,12,13,14,15);
}

// B operand, 32x16 f16 (K x N), rows = N with K contiguous (W[n][k] layout).
// Lane l<16: N=l, K=kofs+0..15;  lane l>=16: N=l-16, K=kofs+16..31.
static __device__ __forceinline__ v16h frag_b(const _Float16* base, int stride,
                                              int lane, int kofs) {
  int n  = lane & 15;
  int kh = (lane >> 4) * 16;
  const _Float16* p = base + n * stride + kofs + kh;
  v8h lo = *(const v8h*)(p);
  v8h hi = *(const v8h*)(p + 8);
  return __builtin_shufflevector(lo, hi, 0,1,2,3,4,5,6,7,8,9,10,11,12,13,14,15);
}

// B operand gathered from a K-major tile (tile[k][n], row stride `stride`):
// used for P x V where V stays in natural [key][d] layout.
static __device__ __forceinline__ v16h frag_b_cm(const _Float16* base, int stride,
                                                 int lane, int kofs, int n0) {
  int n  = n0 + (lane & 15);
  int k0 = kofs + (lane >> 4) * 16;
  v16h r;
#pragma unroll
  for (int j = 0; j < 16; ++j) r[j] = base[(k0 + j) * stride + n];
  return r;
}

static __device__ __forceinline__ v8f wmma_f16(v16h a, v16h b, v8f c) {
  return __builtin_amdgcn_wmma_f32_16x16x32_f16(false, a, false, b,
                                                (short)0, c, false, false);
}

static __device__ __forceinline__ v8h cvt8(float4 a, float4 b) {
  v8h r;
  r[0] = (_Float16)a.x; r[1] = (_Float16)a.y; r[2] = (_Float16)a.z; r[3] = (_Float16)a.w;
  r[4] = (_Float16)b.x; r[5] = (_Float16)b.y; r[6] = (_Float16)b.z; r[7] = (_Float16)b.w;
  return r;
}
static __device__ __forceinline__ v8h cvt8s(float4 a, float4 b, float s) {
  v8h r;
  r[0] = (_Float16)(a.x*s); r[1] = (_Float16)(a.y*s); r[2] = (_Float16)(a.z*s); r[3] = (_Float16)(a.w*s);
  r[4] = (_Float16)(b.x*s); r[5] = (_Float16)(b.y*s); r[6] = (_Float16)(b.z*s); r[7] = (_Float16)(b.w*s);
  return r;
}

// ---------------- generic GEMM: C = A @ W^T + bias (opt ReLU) ----------------
// A: [M,K] f32, W: [N,K] f32, C: [M,N] f32. K % 64 == 0, M % 64 == 0, N % 128 == 0.
// Block: 256 thr (8 waves). C tile 64x128, BK=64. Wave tile 16x64 (4 accs).
__global__ __launch_bounds__(256)
void gemm_wmma(const float* __restrict__ A, const float* __restrict__ W,
               const float* __restrict__ bias, float* __restrict__ C,
               int M, int N, int K, int relu) {
  alignas(16) __shared__ _Float16 As[64][72];
  alignas(16) __shared__ _Float16 Bs[128][72];
  int t = threadIdx.x, lane = t & 31, w = t >> 5;
  int wr = w & 3, wc = w >> 2;
  int m0 = blockIdx.y * 64, n0 = blockIdx.x * 128;
  int ar = t >> 2, ac = (t & 3) * 16;    // A: 64 rows x 64 k, 16 elems/thread
  int br = t >> 1, bc = (t & 1) * 32;    // B: 128 rows x 64 k, 32 elems/thread
  const float* abase = A + (size_t)(m0 + ar) * K + ac;
  const float* wbase = W + (size_t)(n0 + br) * K + bc;
  v8f acc[4] = {};
  for (int kb = 0; kb < K; kb += 64) {
    const float4* ap = (const float4*)(abase + kb);
    const float4* wp = (const float4*)(wbase + kb);
    __builtin_prefetch((const void*)(abase + kb + 64), 0, 0);
    __builtin_prefetch((const void*)(wbase + kb + 64), 0, 0);
    {
      float4 f0 = ap[0], f1 = ap[1], f2 = ap[2], f3 = ap[3];
      *(v8h*)&As[ar][ac]     = cvt8(f0, f1);
      *(v8h*)&As[ar][ac + 8] = cvt8(f2, f3);
    }
#pragma unroll
    for (int q = 0; q < 2; ++q) {
      float4 f0 = wp[4*q+0], f1 = wp[4*q+1], f2 = wp[4*q+2], f3 = wp[4*q+3];
      *(v8h*)&Bs[br][bc + 16*q]     = cvt8(f0, f1);
      *(v8h*)&Bs[br][bc + 16*q + 8] = cvt8(f2, f3);
    }
    __syncthreads();
    {
      // Batch all fragment loads into distinct registers first so the LDS
      // loads can overlap the WMMA pipe (partial dscnt waits, no serialization).
      v16h a0 = frag_a(&As[wr * 16][0], 72, lane, 0);
      v16h a1 = frag_a(&As[wr * 16][0], 72, lane, 32);
      v16h bf0[4], bf1[4];
#pragma unroll
      for (int c = 0; c < 4; ++c) {
        bf0[c] = frag_b(&Bs[wc * 64 + c * 16][0], 72, lane, 0);
        bf1[c] = frag_b(&Bs[wc * 64 + c * 16][0], 72, lane, 32);
      }
#pragma unroll
      for (int c = 0; c < 4; ++c) acc[c] = wmma_f16(a0, bf0[c], acc[c]);
#pragma unroll
      for (int c = 0; c < 4; ++c) acc[c] = wmma_f16(a1, bf1[c], acc[c]);
    }
    __syncthreads();
  }
  int ml = (lane >> 4) * 8, nl = lane & 15;
#pragma unroll
  for (int c = 0; c < 4; ++c) {
    int n = n0 + wc * 64 + c * 16 + nl;
    float bv = bias[n];
#pragma unroll
    for (int i = 0; i < 8; ++i) {
      int m = m0 + wr * 16 + ml + i;
      float v = acc[c][i] + bv;
      if (relu) v = fmaxf(v, 0.0f);
      C[(size_t)m * N + n] = v;
    }
  }
}

// ---------------- flash attention ----------------
// qkv: [B*S, 3*D] f32 (q|k|v), bias: [B,S] per-key, out: [B*S, D] f32.
// Block: 256 thr = 8 waves; each wave owns 16 query rows; tile = 128 q x 64 kv.
__global__ __launch_bounds__(256)
void flash_attn(const float* __restrict__ qkv, const float* __restrict__ bias,
                float* __restrict__ out) {
  alignas(16) __shared__ _Float16 Qs[128][72];
  alignas(16) __shared__ _Float16 Ks[64][72];
  alignas(16) __shared__ _Float16 Vs[64][72];      // natural: [key][d]
  alignas(16) __shared__ _Float16 Ps[8][16][72];   // per-wave P tile
  int t = threadIdx.x, lane = t & 31, w = t >> 5;
  int b = blockIdx.z, hh = blockIdx.y, q0 = blockIdx.x * 128;
  {
    int row = t >> 1, c0 = (t & 1) * 32;
    const float4* src = (const float4*)(qkv + (size_t)(b * S_ + q0 + row) * (3 * D_)
                                        + hh * DH_ + c0);
#pragma unroll
    for (int q = 0; q < 4; ++q)
      *(v8h*)&Qs[row][c0 + 8*q] = cvt8s(src[2*q], src[2*q+1], 0.125f);
  }
  int rowbase = w * 16;
  float mrun[8], lrun[8];
#pragma unroll
  for (int i = 0; i < 8; ++i) { mrun[i] = -1e30f; lrun[i] = 0.0f; }
  v8f o[4] = {};

  for (int j0 = 0; j0 < S_; j0 += 64) {
    __syncthreads();   // prior-iteration readers done (and Q ready on iter 0)
    {
      int row = t >> 2, c0 = (t & 3) * 16;
      size_t rb = (size_t)(b * S_ + j0 + row) * (3 * D_) + hh * DH_ + c0;
      const float4* kp = (const float4*)(qkv + rb + D_);
      const float4* vp = (const float4*)(qkv + rb + 2 * D_);
      *(v8h*)&Ks[row][c0]     = cvt8(kp[0], kp[1]);
      *(v8h*)&Ks[row][c0 + 8] = cvt8(kp[2], kp[3]);
      *(v8h*)&Vs[row][c0]     = cvt8(vp[0], vp[1]);
      *(v8h*)&Vs[row][c0 + 8] = cvt8(vp[2], vp[3]);
    }
    __syncthreads();
    v16h qa0 = frag_a(&Qs[rowbase][0], 72, lane, 0);
    v16h qa1 = frag_a(&Qs[rowbase][0], 72, lane, 32);
    v8f s[4] = {};
    {
      v16h kf[4];
#pragma unroll
      for (int t4 = 0; t4 < 4; ++t4) kf[t4] = frag_b(&Ks[t4 * 16][0], 72, lane, 0);
#pragma unroll
      for (int t4 = 0; t4 < 4; ++t4) s[t4] = wmma_f16(qa0, kf[t4], s[t4]);
#pragma unroll
      for (int t4 = 0; t4 < 4; ++t4) kf[t4] = frag_b(&Ks[t4 * 16][0], 72, lane, 32);
#pragma unroll
      for (int t4 = 0; t4 < 4; ++t4) s[t4] = wmma_f16(qa1, kf[t4], s[t4]);
    }
    float mloc[8];
#pragma unroll
    for (int i = 0; i < 8; ++i) mloc[i] = -3e38f;
#pragma unroll
    for (int t4 = 0; t4 < 4; ++t4) {
      float bv = bias[b * S_ + j0 + t4 * 16 + (lane & 15)];
#pragma unroll
      for (int i = 0; i < 8; ++i) {
        s[t4][i] += bv;
        mloc[i] = fmaxf(mloc[i], s[t4][i]);
      }
    }
#pragma unroll
    for (int msk = 1; msk < 16; msk <<= 1)
#pragma unroll
      for (int i = 0; i < 8; ++i)
        mloc[i] = fmaxf(mloc[i], __shfl_xor(mloc[i], msk, 32));
    float scale[8], rs[8];
#pragma unroll
    for (int i = 0; i < 8; ++i) {
      float mn = fmaxf(mrun[i], mloc[i]);
      scale[i] = __expf(mrun[i] - mn);
      mrun[i]  = mn;
      rs[i]    = 0.0f;
    }
    int prow = 8 * (lane >> 4), pc = lane & 15;
#pragma unroll
    for (int t4 = 0; t4 < 4; ++t4)
#pragma unroll
      for (int i = 0; i < 8; ++i) {
        float p = __expf(s[t4][i] - mrun[i]);
        rs[i] += p;
        Ps[w][prow + i][t4 * 16 + pc] = (_Float16)p;
      }
#pragma unroll
    for (int msk = 1; msk < 16; msk <<= 1)
#pragma unroll
      for (int i = 0; i < 8; ++i) rs[i] += __shfl_xor(rs[i], msk, 32);
#pragma unroll
    for (int i = 0; i < 8; ++i) {
      lrun[i] = lrun[i] * scale[i] + rs[i];
      o[0][i] *= scale[i]; o[1][i] *= scale[i]; o[2][i] *= scale[i]; o[3][i] *= scale[i];
    }
    asm volatile("s_wait_dscnt 0" ::: "memory");   // P tile visible to own wave
    v16h pa0 = frag_a(&Ps[w][0][0], 72, lane, 0);
    v16h pa1 = frag_a(&Ps[w][0][0], 72, lane, 32);
    {
      v16h vf[4];
#pragma unroll
      for (int t4 = 0; t4 < 4; ++t4) vf[t4] = frag_b_cm(&Vs[0][0], 72, lane, 0, t4 * 16);
#pragma unroll
      for (int t4 = 0; t4 < 4; ++t4) o[t4] = wmma_f16(pa0, vf[t4], o[t4]);
#pragma unroll
      for (int t4 = 0; t4 < 4; ++t4) vf[t4] = frag_b_cm(&Vs[0][0], 72, lane, 32, t4 * 16);
#pragma unroll
      for (int t4 = 0; t4 < 4; ++t4) o[t4] = wmma_f16(pa1, vf[t4], o[t4]);
    }
  }
#pragma unroll
  for (int i = 0; i < 8; ++i) {
    int r = q0 + rowbase + 8 * (lane >> 4) + i;
    float inv = 1.0f / lrun[i];
    size_t base = (size_t)(b * S_ + r) * D_ + hh * DH_ + (lane & 15);
    out[base +  0] = o[0][i] * inv;
    out[base + 16] = o[1][i] * inv;
    out[base + 32] = o[2][i] * inv;
    out[base + 48] = o[3][i] * inv;
  }
}

// ---------------- residual + LayerNorm (in place on h) ----------------
__global__ __launch_bounds__(256)
void ln_residual(float* __restrict__ h, const float* __restrict__ r,
                 const float* __restrict__ g, const float* __restrict__ bta) {
  int lane = threadIdx.x & 31, w = threadIdx.x >> 5;
  int row  = blockIdx.x * 8 + w;
  size_t base = (size_t)row * D_;
  float x[16];
  float s = 0.0f;
#pragma unroll
  for (int j = 0; j < 16; ++j) {
    int c = lane + j * 32;
    x[j] = h[base + c] + r[base + c];
    s += x[j];
  }
#pragma unroll
  for (int m = 1; m < 32; m <<= 1) s += __shfl_xor(s, m, 32);
  float mean = s * (1.0f / D_);
  float v = 0.0f;
#pragma unroll
  for (int j = 0; j < 16; ++j) { float d = x[j] - mean; v += d * d; }
#pragma unroll
  for (int m = 1; m < 32; m <<= 1) v += __shfl_xor(v, m, 32);
  float rstd = rsqrtf(v * (1.0f / D_) + 1e-5f);
#pragma unroll
  for (int j = 0; j < 16; ++j) {
    int c = lane + j * 32;
    h[base + c] = (x[j] - mean) * rstd * g[c] + bta[c];
  }
}

// ---------------- pad / attention bias + target-nonzero flags ----------------
__global__ __launch_bounds__(256)
void pad_bias(const float* __restrict__ xc, const float* __restrict__ xt,
              float* __restrict__ bias, float* __restrict__ tgtnz) {
  int lane = threadIdx.x & 31, w = threadIdx.x >> 5;
  int row = blockIdx.x * 8 + w;         // 0..B*S-1
  int b = row / S_, s = row % S_;
  const float* src = (s < NC_) ? xc + (size_t)(b * NC_ + s) * DX_
                               : xt + (size_t)(b * NT_ + (s - NC_)) * DX_;
  float sum = 0.0f;
#pragma unroll
  for (int j = 0; j < 32; ++j) sum += src[lane + j * 32];
#pragma unroll
  for (int m = 1; m < 32; m <<= 1) sum += __shfl_xor(sum, m, 32);
  if (lane == 0) {
    int nz = (sum != 0.0f);
    bias[row] = (s < NC_ && nz) ? 0.0f : NEGB;
    if (s >= NC_) tgtnz[b * NT_ + (s - NC_)] = nz ? 1.0f : 0.0f;
  }
}

// ---------------- input concat, zero-padded to KP_ ----------------
__global__ void build_inp(const float* __restrict__ xc, const float* __restrict__ yc,
                          const float* __restrict__ xt, float* __restrict__ inp) {
  const int total = B_ * S_ * KP_;
  for (int idx = blockIdx.x * blockDim.x + threadIdx.x; idx < total;
       idx += gridDim.x * blockDim.x) {
    int b = idx / (S_ * KP_);
    int rem = idx % (S_ * KP_);
    int s = rem / KP_, d = rem % KP_;
    float v = 0.0f;
    if (s < NC_) {
      if (d < DX_)        v = xc[(size_t)(b * NC_ + s) * DX_ + d];
      else if (d == DX_)  v = yc[b * NC_ + s];
    } else {
      if (d < DX_)        v = xt[(size_t)(b * NT_ + (s - NC_)) * DX_ + d];
    }
    inp[idx] = v;
  }
}

// ---------------- zero-pad emb_w1 [D,KIN] -> [D,KP] ----------------
__global__ void pad_w1(const float* __restrict__ w1, float* __restrict__ out) {
  const int total = D_ * KP_;
  for (int idx = blockIdx.x * blockDim.x + threadIdx.x; idx < total;
       idx += gridDim.x * blockDim.x) {
    int n = idx / KP_, k = idx % KP_;
    out[idx] = (k < KIN_) ? w1[(size_t)n * KIN_ + k] : 0.0f;
  }
}

// ---------------- copy target rows of h -> z ----------------
__global__ void copy_z(const float* __restrict__ h, float* __restrict__ z) {
  const int total = B_ * NT_ * D_;
  for (int idx = blockIdx.x * blockDim.x + threadIdx.x; idx < total;
       idx += gridDim.x * blockDim.x) {
    int b = idx / (NT_ * D_);
    int rem = idx % (NT_ * D_);
    int tt = rem / D_, d = rem % D_;
    z[idx] = h[(size_t)(b * S_ + NC_ + tt) * D_ + d];
  }
}

// ---------------- prediction head 2: [M,DFF] x [2,DFF]^T ----------------
__global__ __launch_bounds__(256)
void head2(const float* __restrict__ p1, const float* __restrict__ w2,
           const float* __restrict__ b2, float* __restrict__ mraw,
           float* __restrict__ sraw) {
  int lane = threadIdx.x & 31, w = threadIdx.x >> 5;
  int row = blockIdx.x * 8 + w;          // 0..B*NT-1
  const float* pr = p1 + (size_t)row * DFF_;
  float s0 = 0.0f, s1 = 0.0f;
#pragma unroll
  for (int j = 0; j < 64; ++j) {
    int c = lane + j * 32;
    float x = pr[c];
    s0 += x * w2[c];
    s1 += x * w2[DFF_ + c];
  }
#pragma unroll
  for (int m = 1; m < 32; m <<= 1) {
    s0 += __shfl_xor(s0, m, 32);
    s1 += __shfl_xor(s1, m, 32);
  }
  if (lane == 0) { mraw[row] = s0 + b2[0]; sraw[row] = s1 + b2[1]; }
}

// ---------------- segment softmax + Gaussian LL (one block per batch) ------
__global__ __launch_bounds__(256)
void seg_ll(const int* __restrict__ tarqlen, const float* __restrict__ yt,
            const float* __restrict__ mraw, const float* __restrict__ sraw,
            const float* __restrict__ tgtnz, float* __restrict__ partial) {
  __shared__ int csum[NSEG_];
  __shared__ float red[256];
  int b = blockIdx.x, t = threadIdx.x;
  if (t == 0) {
    int run = 0;
    for (int i = 0; i < NSEG_; ++i) { run += tarqlen[b * NSEG_ + i]; csum[i] = run; }
  }
  float c = 0.0f;
  for (int i = t; i < NT_; i += 256) c += tgtnz[b * NT_ + i];
  red[t] = c;
  __syncthreads();
  for (int st = 128; st > 0; st >>= 1) { if (t < st) red[t] += red[t + st]; __syncthreads(); }
  int lens = (int)(red[0] + 0.5f);
  __syncthreads();
  float acc = 0.0f;
  if (t < NSEG_) {
    int start = (t == 0) ? 0 : csum[t - 1];
    int end = csum[t];
    if (end > start) {
      float m = -3e38f;
      for (int i = start; i < end; ++i) m = fmaxf(m, mraw[b * NT_ + i]);
      float ssum = 0.0f;
      for (int i = start; i < end; ++i) ssum += __expf(mraw[b * NT_ + i] - m);
      float inv = 1.0f / ssum;
      for (int i = start; i < end; ++i) {
        float mean = __expf(mraw[b * NT_ + i] - m) * inv;
        float sr = sraw[b * NT_ + i];
        float istd = __expf(-sr);
        float diff = (yt[b * NT_ + i] - mean) * istd;
        float lp = -0.5f * LN2PI_ - sr - 0.5f * diff * diff;
        if (i < lens) acc += lp;
      }
    }
  }
  red[t] = acc;
  __syncthreads();
  for (int st = 128; st > 0; st >>= 1) { if (t < st) red[t] += red[t + st]; __syncthreads(); }
  if (t == 0) partial[b] = red[0];
}

__global__ void finalize_k(const float* __restrict__ partial, float* __restrict__ out) {
  float s = 0.0f;
  for (int i = 0; i < B_; ++i) s += partial[i];
  float ll = s / (float)(B_ * NT_);
  out[0] = ll;
  out[1] = -ll;
}

// ---------------- driver ----------------
extern "C" void kernel_launch(void* const* d_in, const int* in_sizes, int n_in,
                              void* d_out, int out_size, void* d_ws, size_t ws_size,
                              hipStream_t stream) {
  (void)in_sizes; (void)n_in; (void)out_size; (void)ws_size;
  const float* xc     = (const float*)d_in[0];
  const float* yc     = (const float*)d_in[1];
  const float* xt     = (const float*)d_in[2];
  const float* yt     = (const float*)d_in[3];
  const int*   tarqlen= (const int*)  d_in[4];
  const float* emb_w1 = (const float*)d_in[5];
  const float* emb_b1 = (const float*)d_in[6];
  const float* emb_w2 = (const float*)d_in[7];
  const float* emb_b2 = (const float*)d_in[8];
  const float* qkv_w  = (const float*)d_in[9];
  const float* qkv_b  = (const float*)d_in[10];
  const float* ao_w   = (const float*)d_in[11];
  const float* ao_b   = (const float*)d_in[12];
  const float* ff1_w  = (const float*)d_in[13];
  const float* ff1_b  = (const float*)d_in[14];
  const float* ff2_w  = (const float*)d_in[15];
  const float* ff2_b  = (const float*)d_in[16];
  const float* ln1_g  = (const float*)d_in[17];
  const float* ln1_b  = (const float*)d_in[18];
  const float* ln2_g  = (const float*)d_in[19];
  const float* ln2_b  = (const float*)d_in[20];
  const float* pr_w1  = (const float*)d_in[21];
  const float* pr_b1  = (const float*)d_in[22];
  const float* pr_w2  = (const float*)d_in[23];
  const float* pr_b2  = (const float*)d_in[24];

  char* wsb = (char*)d_ws;
  size_t off = 0;
  auto alloc = [&](size_t nflt) {
    float* p = (float*)(wsb + off);
    off += ((nflt * sizeof(float) + 255) / 256) * 256;
    return p;
  };
  const size_t MS = (size_t)B_ * S_;       // 8192
  float* R1    = alloc(MS * DFF_);         // inp (8192xKP) / ff1 out / pred p1
  float* h     = alloc(MS * D_);
  float* qkvb  = alloc(MS * 3 * D_);
  float* abuf  = alloc(MS * D_);
  float* tbuf  = alloc(MS * D_);
  float* biasb = alloc(MS);
  float* tgtnz = alloc((size_t)B_ * NT_);
  float* z     = alloc((size_t)B_ * NT_ * D_);
  float* w1p   = alloc((size_t)D_ * KP_);
  float* mraw  = alloc((size_t)B_ * NT_);
  float* sraw  = alloc((size_t)B_ * NT_);
  float* partial = alloc(B_);

  auto gemm = [&](const float* A, const float* W, const float* bias, float* C,
                  int M, int N, int K, int relu) {
    dim3 g(N / 128, M / 64);
    gemm_wmma<<<g, 256, 0, stream>>>(A, W, bias, C, M, N, K, relu);
  };

  // --- embedding ---
  build_inp<<<2048, 256, 0, stream>>>(xc, yc, xt, R1);
  pad_w1<<<512, 256, 0, stream>>>(emb_w1, w1p);
  pad_bias<<<(int)(MS / 8), 256, 0, stream>>>(xc, xt, biasb, tgtnz);
  gemm(R1, w1p, emb_b1, tbuf, (int)MS, D_, KP_, 1);
  gemm(tbuf, emb_w2, emb_b2, h, (int)MS, D_, D_, 0);

  // --- transformer layers ---
  for (int l = 0; l < L_; ++l) {
    gemm(h, qkv_w + (size_t)l * 3 * D_ * D_, qkv_b + (size_t)l * 3 * D_, qkvb,
         (int)MS, 3 * D_, D_, 0);
    flash_attn<<<dim3(S_ / 128, H_, B_), 256, 0, stream>>>(qkvb, biasb, abuf);
    gemm(abuf, ao_w + (size_t)l * D_ * D_, ao_b + (size_t)l * D_, tbuf,
         (int)MS, D_, D_, 0);
    ln_residual<<<(int)(MS / 8), 256, 0, stream>>>(h, tbuf,
         ln1_g + (size_t)l * D_, ln1_b + (size_t)l * D_);
    gemm(h, ff1_w + (size_t)l * DFF_ * D_, ff1_b + (size_t)l * DFF_, R1,
         (int)MS, DFF_, D_, 1);
    gemm(R1, ff2_w + (size_t)l * D_ * DFF_, ff2_b + (size_t)l * D_, tbuf,
         (int)MS, D_, DFF_, 0);
    ln_residual<<<(int)(MS / 8), 256, 0, stream>>>(h, tbuf,
         ln2_g + (size_t)l * D_, ln2_b + (size_t)l * D_);
  }

  // --- prediction head ---
  copy_z<<<2048, 256, 0, stream>>>(h, z);
  gemm(z, pr_w1, pr_b1, R1, B_ * NT_, DFF_, D_, 1);
  head2<<<(B_ * NT_) / 8, 256, 0, stream>>>(R1, pr_w2, pr_b2, mraw, sraw);

  // --- segment softmax + log-likelihood ---
  seg_ll<<<B_, 256, 0, stream>>>(tarqlen, yt, mraw, sraw, tgtnz, partial);
  finalize_k<<<1, 1, 0, stream>>>(partial, (float*)d_out);
}
